// PRCNN_71528385348277
// MI455X (gfx1250) — matrix operand order repaired and checked by reading, not compiled, not run on hardware
//
#include <hip/hip_runtime.h>

typedef __attribute__((ext_vector_type(16))) _Float16 v16h;
typedef __attribute__((ext_vector_type(8)))  float    v8f;

#define NEGV (-1e30f)

// geometry
#define BATCH 16
#define H0 384
#define W0 512
#define C1 10
#define H1 191
#define W1 255
#define C2 16
#define H2 189
#define W2 253
#define NP2 (H2*W2)        // 47817
#define T2  ((NP2+15)/16)  // 2989 tiles/image
#define K2  90
#define C3 32
#define H3 187
#define W3 251
#define NP3 (H3*W3)        // 46937
#define T3  ((NP3+15)/16)  // 2934 tiles/image
#define K3  144
#define LSTRIDE (T3*16)    // 46944 padded live stride
#define KP 128

// ---------------- conv1 (3->10, 3x3) + prelu + 2x2 maxpool, fused -------------
// output stored as f16 (it is only ever consumed as a WMMA f16 operand)
__global__ void k_conv1_pool(const float* __restrict__ im, const float* __restrict__ w,
                             const float* __restrict__ bias, const float* __restrict__ pr,
                             _Float16* __restrict__ x1) {
  int t = blockIdx.x * blockDim.x + threadIdx.x;
  int total = BATCH*C1*H1*W1;
  if (t >= total) return;
  int px = t % W1; int r = t / W1;
  int py = r % H1; r /= H1;
  int oc = r % C1; int b = r / C1;
  float s00 = bias[oc], s01 = s00, s10 = s00, s11 = s00;
  int oy = 2*py, ox = 2*px;
  for (int c = 0; c < 3; ++c) {
    const float* ip = im + ((size_t)(b*3 + c)*H0 + oy)*W0 + ox;
    const float* wp = w + (oc*3 + c)*9;
    #pragma unroll
    for (int ky = 0; ky < 3; ++ky) {
      #pragma unroll
      for (int kx = 0; kx < 3; ++kx) {
        float wv = wp[ky*3+kx];
        float i00 = (ip[(ky+0)*W0 + kx+0] - 127.5f)*0.0078125f;
        float i01 = (ip[(ky+0)*W0 + kx+1] - 127.5f)*0.0078125f;
        float i10 = (ip[(ky+1)*W0 + kx+0] - 127.5f)*0.0078125f;
        float i11 = (ip[(ky+1)*W0 + kx+1] - 127.5f)*0.0078125f;
        s00 += wv*i00; s01 += wv*i01; s10 += wv*i10; s11 += wv*i11;
      }
    }
  }
  float a = pr[oc];
  s00 = s00>0.f?s00:a*s00; s01 = s01>0.f?s01:a*s01;
  s10 = s10>0.f?s10:a*s10; s11 = s11>0.f?s11:a*s11;
  x1[t] = (_Float16)fmaxf(fmaxf(s00,s01), fmaxf(s10,s11));
}

// ---------------- pack conv2/conv3 weights into WMMA B-fragment layout (f16) --
// B fragment element mapping (32x16 f16 B): lane<16 -> N=lane, K=chunk*32+(e<8?e:e+8)
//                                           lane>=16 -> N=lane-16, K=chunk*32+8+(e<8?e:e+8)
// K-padding entries are exact 0.0 => A values at padded k are don't-care.
__global__ void k_pack(const float* __restrict__ w2, const float* __restrict__ w3,
                       _Float16* __restrict__ pb2, _Float16* __restrict__ pb3) {
  int t = threadIdx.x + blockIdx.x*blockDim.x;
  int stride = blockDim.x * gridDim.x;
  // conv2: [chunk(3)][lane(32)][e(16)]
  for (int i = t; i < 3*32*16; i += stride) {
    int e = i & 15; int lane = (i >> 4) & 31; int chunk = i >> 9;
    int n = lane & 15;
    int k = chunk*32 + (e < 8 ? e : e + 8) + ((lane < 16) ? 0 : 8);
    float v = 0.f;
    if (k < K2) {
      int ic = k/9, off = k%9, ky = off/3, kx = off - ky*3;
      v = w2[((n*C1 + ic)*3 + ky)*3 + kx];
    }
    pb2[i] = (_Float16)v;
  }
  // conv3: [tile(2)][chunk(5)][lane(32)][e(16)]
  for (int i = t; i < 2*5*32*16; i += stride) {
    int e = i & 15; int lane = (i >> 4) & 31; int chunk = (i >> 9) % 5; int tile = i / 2560;
    int n = (lane & 15) + tile*16;
    int k = chunk*32 + (e < 8 ? e : e + 8) + ((lane < 16) ? 0 : 8);
    float v = 0.f;
    if (k < K3) {
      int ic = k/9, off = k%9, ky = off/3, kx = off - ky*3;
      v = w3[((n*C2 + ic)*3 + ky)*3 + kx];
    }
    pb3[i] = (_Float16)v;
  }
}

// ---------------- A-fragment gather (implicit im2col), select-free ------------
// 16-bit A layout: lanes 0-15 hold K 0-7,16-23 ; lanes 16-31 hold K 8-15,24-31
// => element e of lane group hi maps to k = CHUNK*32 + (e<8 ? e : e+8) + 8*hi.
// No value masking is needed: padded-k B weights are 0.0 and out-of-range tile
// rows are never stored, so only the ADDRESS must stay in-bounds (clamped k and
// caller-clamped y/x). All loads are unconditional -> the compiler can clause
// them and overlap waits with the WMMA chain.
template<int CHUNK, int K, int IC, int HIN, int WIN>
__device__ __forceinline__ v16h build_a(const _Float16* __restrict__ img, // + b*IC*HIN*WIN
                                        int hi, int y, int x) {
  v16h a;
  #pragma unroll
  for (int e = 0; e < 16; ++e) {
    const int k0 = CHUNK*32 + (e < 8 ? e : e + 8);   // compile-time
    if (k0 >= K) {
      a[e] = (_Float16)0.f;            // padding for both halves: no load at all
    } else {
      int k = k0 + 8*hi;
      if (k0 + 8 >= K) {               // compile-time: clamp only boundary elems
        k = (k < K) ? k : (K - 1);
      }
      int ic = k / 9;
      int off = k - ic*9;
      int ky = off / 3;
      int kx = off - ky*3;
      a[e] = img[(size_t)(ic*HIN + (y+ky))*WIN + (x+kx)];
    }
  }
  return a;
}

// ---------------- conv2 (10->16, 3x3) + prelu via WMMA implicit GEMM ----------
__global__ void k_conv2(const _Float16* __restrict__ x1, const _Float16* __restrict__ pbw,
                        const float* __restrict__ bias, const float* __restrict__ pr,
                        _Float16* __restrict__ h2) {
  int wave = (blockIdx.x * blockDim.x + threadIdx.x) >> 5;
  int lane = threadIdx.x & 31;
  if (wave >= BATCH*T2) return;
  int b = wave / T2;
  int base = (wave % T2) * 16;
  int p = base + (lane & 15);
  if (p >= NP2) p = NP2 - 1;           // clamp: garbage rows are never stored
  int y = p / W2;
  int x = p % W2;
  int hi = lane >> 4;
  const _Float16* img = x1 + (size_t)b*C1*H1*W1;
  v8f acc = {};
  {
    v16h a0 = build_a<0, K2, C1, H1, W1>(img, hi, y, x);
    v16h bf = *(const v16h*)(pbw + (0*32 + lane)*16);
    acc = __builtin_amdgcn_wmma_f32_16x16x32_f16(false, a0, false, bf, (short)0, acc, false, false);
  }
  {
    v16h a1 = build_a<1, K2, C1, H1, W1>(img, hi, y, x);
    v16h bf = *(const v16h*)(pbw + (1*32 + lane)*16);
    acc = __builtin_amdgcn_wmma_f32_16x16x32_f16(false, a1, false, bf, (short)0, acc, false, false);
  }
  {
    v16h a2 = build_a<2, K2, C1, H1, W1>(img, hi, y, x);
    v16h bf = *(const v16h*)(pbw + (2*32 + lane)*16);
    acc = __builtin_amdgcn_wmma_f32_16x16x32_f16(false, a2, false, bf, (short)0, acc, false, false);
  }
  int n = lane & 15;
  float bb = bias[n], aa = pr[n];
  #pragma unroll
  for (int v = 0; v < 8; ++v) {
    int pp = base + v + 8*hi;         // D layout: VGPR v -> M=v (lanes 0-15) / M=v+8
    if (pp < NP2) {
      float s = acc[v] + bb;
      s = s > 0.f ? s : aa*s;
      int yy = pp / W2, xx = pp % W2;
      h2[((size_t)(b*C2 + n)*H2 + yy)*W2 + xx] = (_Float16)s;
    }
  }
}

// ------- conv3 (16->32, 3x3) + prelu + 1x1 heads + softmax + mask, fused ------
__global__ void k_conv3_head(const _Float16* __restrict__ h2, const _Float16* __restrict__ pbw,
                             const float* __restrict__ bias, const float* __restrict__ pr,
                             const float* __restrict__ w41, const float* __restrict__ b41,
                             const float* __restrict__ w42, const float* __restrict__ b42,
                             float* __restrict__ probs, float* __restrict__ live,
                             float* __restrict__ reg) {
  int wave = (blockIdx.x * blockDim.x + threadIdx.x) >> 5;
  int lane = threadIdx.x & 31;
  if (wave >= BATCH*T3) return;
  int b = wave / T3;
  int base = (wave % T3) * 16;
  int p = base + (lane & 15);
  if (p >= NP3) p = NP3 - 1;           // clamp: garbage rows are never stored
  int y = p / W3;
  int x = p % W3;
  int hi = lane >> 4;
  const _Float16* img = h2 + (size_t)b*C2*H2*W2;
  v8f acc0 = {}, acc1 = {};
  #define C3_STEP(CH)                                                                              \
  {                                                                                                \
    v16h a = build_a<CH, K3, C2, H2, W2>(img, hi, y, x);                                           \
    v16h bf0 = *(const v16h*)(pbw + ((0*5 + CH)*32 + lane)*16);                                    \
    v16h bf1 = *(const v16h*)(pbw + ((1*5 + CH)*32 + lane)*16);                                    \
    acc0 = __builtin_amdgcn_wmma_f32_16x16x32_f16(false, a, false, bf0, (short)0, acc0, false, false); \
    acc1 = __builtin_amdgcn_wmma_f32_16x16x32_f16(false, a, false, bf1, (short)0, acc1, false, false); \
  }
  C3_STEP(0) C3_STEP(1) C3_STEP(2) C3_STEP(3) C3_STEP(4)
  #undef C3_STEP
  int c = lane & 15;                 // this lane's channel (and c+16 in acc1)
  float biasA = bias[c],    aA = pr[c];
  float biasB = bias[c+16], aB = pr[c+16];
  float wA0 = w41[c],     wA0b = w41[16+c];
  float wA1 = w41[32+c],  wA1b = w41[48+c];
  float wr0 = w42[c],     wrb0 = w42[16+c];
  float wr1 = w42[32+c],  wrb1 = w42[48+c];
  float wr2 = w42[64+c],  wrb2 = w42[80+c];
  float wr3 = w42[96+c],  wrb3 = w42[112+c];
  float bz0 = b41[0], bz1 = b41[1];
  float br0 = b42[0], br1 = b42[1], br2 = b42[2], br3 = b42[3];
  #pragma unroll
  for (int v = 0; v < 8; ++v) {
    float hA = acc0[v] + biasA; hA = hA > 0.f ? hA : aA*hA;
    float hB = acc1[v] + biasB; hB = hB > 0.f ? hB : aB*hB;
    float s0 = wA0*hA + wA0b*hB;
    float s1 = wA1*hA + wA1b*hB;
    float r0 = wr0*hA + wrb0*hB;
    float r1 = wr1*hA + wrb1*hB;
    float r2 = wr2*hA + wrb2*hB;
    float r3 = wr3*hA + wrb3*hB;
    #pragma unroll
    for (int off = 8; off >= 1; off >>= 1) {  // reduce within 16-lane halves
      s0 += __shfl_xor(s0, off, 32);
      s1 += __shfl_xor(s1, off, 32);
      r0 += __shfl_xor(r0, off, 32);
      r1 += __shfl_xor(r1, off, 32);
      r2 += __shfl_xor(r2, off, 32);
      r3 += __shfl_xor(r3, off, 32);
    }
    if (c == 0) {
      int pp = base + v + 8*hi;
      if (pp < NP3) {
        float z = (s0 + bz0) - (s1 + bz1);
        float psc = 1.f / (1.f + __expf(z));   // softmax channel 1
        probs[(size_t)b*NP3 + pp] = psc;
        live[(size_t)b*LSTRIDE + pp] = (psc >= 0.6f) ? psc : NEGV;
        float* rp = reg + ((size_t)b*NP3 + pp)*4;
        rp[0] = r0 + br0; rp[1] = r1 + br1; rp[2] = r2 + br2; rp[3] = r3 + br3;
      }
    }
  }
}

// ---------------- greedy NMS: 128 sequential block-wide argmax picks ----------
// boxes are constant 12x12 on a stride-2 grid => iou>0.5 <=> inter>96, which
// only happens for |dx|<=1 && |dy|<=1; suppress a 5x5 superset with exact test.
__global__ void __launch_bounds__(1024) k_nms(float* __restrict__ live,
                                              int* __restrict__ picks,
                                              int* __restrict__ oks) {
  __shared__ float sv[1024];
  __shared__ int   si[1024];
  int b = blockIdx.x;
  int tid = threadIdx.x;
  float* L = live + (size_t)b * LSTRIDE;
  for (int it = 0; it < KP; ++it) {
    float bv = -3.0e38f;
    int bi = 0;
    for (int i = tid; i < NP3; i += 1024) {
      float v = L[i];
      if (v > bv) { bv = v; bi = i; }   // strict '>' keeps first index in-thread
    }
    sv[tid] = bv; si[tid] = bi;
    __syncthreads();
    for (int s = 512; s > 0; s >>= 1) {
      if (tid < s) {
        float ov = sv[tid+s]; int oi = si[tid+s];
        if (ov > sv[tid] || (ov == sv[tid] && oi < si[tid])) { sv[tid] = ov; si[tid] = oi; }
      }
      __syncthreads();
    }
    int j = si[0];
    if (tid == 0) {
      picks[b*KP + it] = j;
      oks[b*KP + it] = (sv[0] > -5e29f) ? 1 : 0;
    }
    if (tid < 25) {
      int jy = j / W3, jx = j % W3;
      int dy = tid/5 - 2, dx = tid%5 - 2;
      int ny = jy + dy, nx = jx + dx;
      if (ny >= 0 && ny < H3 && nx >= 0 && nx < W3) {
        int ady = dy < 0 ? -dy : dy;
        int adx = dx < 0 ? -dx : dx;
        int iw = 12 - 2*adx; if (iw < 0) iw = 0;
        int ih = 12 - 2*ady; if (ih < 0) ih = 0;
        if (iw*ih > 96 || (dx == 0 && dy == 0)) L[ny*W3 + nx] = NEGV;
      }
    }
    __syncthreads();
  }
}

// ---------------- box refinement + output assembly ----------------------------
__global__ void k_final(const int* __restrict__ picks, const int* __restrict__ oks,
                        const float* __restrict__ probs, const float* __restrict__ reg,
                        float* __restrict__ out) {
  int t = blockIdx.x*blockDim.x + threadIdx.x;
  if (t >= BATCH*KP) return;
  int b = t / KP;
  float* o = out + (size_t)t * 5;
  if (!oks[t]) { o[0]=0.f; o[1]=0.f; o[2]=0.f; o[3]=0.f; o[4]=0.f; return; }
  int j = picks[t];
  int iy = j / W3, ix = j % W3;
  float x1 = 2.f*ix + 1.f,  y1 = 2.f*iy + 1.f;
  float x2 = 2.f*ix + 12.f, y2 = 2.f*iy + 12.f;
  const float* r = reg + ((size_t)b*NP3 + j)*4;
  float w = x2 - x1, hh = y2 - y1;
  float q1 = x1 + r[0]*w,  q2 = y1 + r[1]*hh;
  float q3 = x2 + r[2]*w,  q4 = y2 + r[3]*hh;
  float rw = q3 - q1, rh = q4 - q2;
  float l = fmaxf(rw, rh);
  float nx1 = q1 + rw*0.5f - l*0.5f;
  float ny1 = q2 + rh*0.5f - l*0.5f;
  o[0] = nx1; o[1] = ny1; o[2] = nx1 + l; o[3] = ny1 + l;
  o[4] = probs[(size_t)b*NP3 + j];
}

extern "C" void kernel_launch(void* const* d_in, const int* in_sizes, int n_in,
                              void* d_out, int out_size, void* d_ws, size_t ws_size,
                              hipStream_t stream) {
  (void)in_sizes; (void)n_in; (void)out_size; (void)ws_size;
  const float* im   = (const float*)d_in[0];
  const float* c1w  = (const float*)d_in[1];
  const float* c1b  = (const float*)d_in[2];
  const float* p1   = (const float*)d_in[3];
  const float* c2w  = (const float*)d_in[4];
  const float* c2b  = (const float*)d_in[5];
  const float* p2   = (const float*)d_in[6];
  const float* c3w  = (const float*)d_in[7];
  const float* c3b  = (const float*)d_in[8];
  const float* p3   = (const float*)d_in[9];
  const float* c41w = (const float*)d_in[10];
  const float* c41b = (const float*)d_in[11];
  const float* c42w = (const float*)d_in[12];
  const float* c42b = (const float*)d_in[13];

  char* wsb = (char*)d_ws;
  size_t off = 0;
  auto take = [&](size_t bytes) -> char* {
    char* p = wsb + off;
    off += (bytes + 255) & ~(size_t)255;
    return p;
  };
  _Float16* x1    = (_Float16*)take((size_t)BATCH*C1*H1*W1*2);
  _Float16* h2    = (_Float16*)take((size_t)BATCH*C2*H2*W2*2);
  float*    probs = (float*)take((size_t)BATCH*NP3*4);
  float*    live  = (float*)take((size_t)BATCH*LSTRIDE*4);
  float*    reg   = (float*)take((size_t)BATCH*NP3*4*4);
  _Float16* pb2   = (_Float16*)take((size_t)3*32*16*2);
  _Float16* pb3   = (_Float16*)take((size_t)2*5*32*16*2);
  int*      picks = (int*)take((size_t)BATCH*KP*4);
  int*      oks   = (int*)take((size_t)BATCH*KP*4);

  {
    int total = BATCH*C1*H1*W1;
    k_conv1_pool<<<(total+255)/256, 256, 0, stream>>>(im, c1w, c1b, p1, x1);
  }
  k_pack<<<4, 256, 0, stream>>>(c2w, c3w, pb2, pb3);
  {
    int waves = BATCH*T2;
    k_conv2<<<(waves+7)/8, 256, 0, stream>>>(x1, pb2, c2b, p2, h2);
  }
  {
    int waves = BATCH*T3;
    k_conv3_head<<<(waves+7)/8, 256, 0, stream>>>(h2, pb3, c3b, p3,
                                                  c41w, c41b, c42w, c42b,
                                                  probs, live, reg);
  }
  k_nms<<<BATCH, 1024, 0, stream>>>(live, picks, oks);
  k_final<<<(BATCH*KP+255)/256, 256, 0, stream>>>(picks, oks, probs, reg, (float*)d_out);
}
